// M4C_84808424227272
// MI455X (gfx1250) — matrix operand order repaired and checked by reading, not compiled, not run on hardware
//
#include <hip/hip_runtime.h>
#include <hip/hip_bf16.h>

typedef __attribute__((ext_vector_type(16))) _Float16 v16h;
typedef __attribute__((ext_vector_type(8)))  _Float16 v8h;
typedef __attribute__((ext_vector_type(4)))  _Float16 v4h;
typedef __attribute__((ext_vector_type(8)))  float    v8f;

#define B_   64
#define N_   182
#define NP   192      // N padded to 12 tiles of 16
#define H_   768
#define NH_  12
#define HD_  64
#define MSL_ 20
#define OCR_ 150

static __device__ __forceinline__ v16h join8(v8h lo, v8h hi) {
    return __builtin_shufflevector(lo, hi, 0, 1, 2, 3, 4, 5, 6, 7,
                                   8, 9, 10, 11, 12, 13, 14, 15);
}

// ---------------------------------------------------------------------------
// Zero-fill padded rows/cols of the f16 Q/K/V^T workspaces so that WMMA never
// multiplies uninitialized data (0 * Inf = NaN hazard per ISA).
// ---------------------------------------------------------------------------
__global__ void zero_pad_kernel(_Float16* __restrict__ qws,
                                _Float16* __restrict__ kws,
                                _Float16* __restrict__ vT) {
    const int tid = blockIdx.x * blockDim.x + threadIdx.x;
    const int per = (NP - N_) * HD_;                  // 640 per (b,h)
    const int total = B_ * NH_ * per;                 // 491520
    if (tid >= total) return;
    const int bh  = tid / per;
    const int rem = tid % per;
    const int n = N_ + rem / HD_;
    const int d = rem % HD_;
    qws[((size_t)bh * NP + n) * HD_ + d] = (_Float16)0.f;
    kws[((size_t)bh * NP + n) * HD_ + d] = (_Float16)0.f;
    const int d2 = rem / (NP - N_);
    const int n2 = N_ + rem % (NP - N_);
    vT[((size_t)bh * HD_ + d2) * NP + n2] = (_Float16)0.f;
}

// ---------------------------------------------------------------------------
// One-shot f32 -> f16 conversion (4 elements / thread, b128 in, b64 out).
// ---------------------------------------------------------------------------
__global__ void cvt_f32_to_f16_kernel(const float* __restrict__ src,
                                      _Float16* __restrict__ dst, int n4) {
    const int i = blockIdx.x * blockDim.x + threadIdx.x;
    if (i >= n4) return;
    const float4 v = ((const float4*)src)[i];
    v4h o;
    o[0] = (_Float16)v.x; o[1] = (_Float16)v.y;
    o[2] = (_Float16)v.z; o[3] = (_Float16)v.w;
    ((v4h*)dst)[i] = o;
}

// ---------------------------------------------------------------------------
// Fused QKV projection on pre-converted f16 operands: Y = X @ W^T + b.
// One wave computes a 16x64 tile of the (11648 x 2304) fused output.
// Tasks: 728 row-tiles * 36 col-groups = 26208 waves = 3276 blocks of 8.
// Q,K stored f16 [b][h][NP][HD]; V stored transposed f16 [b][h][HD][NP].
// ---------------------------------------------------------------------------
__global__ __launch_bounds__(256) void qkv_proj_kernel(
    const _Float16* __restrict__ Xh,      // (11648, 768) f16
    const _Float16* __restrict__ Wh,      // (3, 768, 768) f16  (Wq|Wk|Wv)
    const float* __restrict__ bq, const float* __restrict__ bk,
    const float* __restrict__ bv,
    _Float16* __restrict__ qws, _Float16* __restrict__ kws,
    _Float16* __restrict__ vT)
{
    const int lane   = threadIdx.x & 31;
    const int wvid   = threadIdx.x >> 5;
    const int task   = blockIdx.x * 8 + wvid;   // < 26208
    const int mtile  = task % 728;
    const int g      = task / 728;              // 0..35
    const int mat    = g / NH_;                 // 0=Q 1=K 2=V
    const int h      = g % NH_;
    const int lanelo = lane & 15;
    const int hi     = lane >> 4;

    const _Float16* Wm   = Wh + (size_t)mat * (H_ * H_);
    const float*    bias = (mat == 0) ? bq : (mat == 1) ? bk : bv;

    v8f acc[4];
    #pragma unroll
    for (int t = 0; t < 4; ++t)
        #pragma unroll
        for (int i = 0; i < 8; ++i) acc[t][i] = 0.f;

    const int arow = mtile * 16 + lanelo;               // A-matrix: lane = M
    const _Float16* xrow = Xh + (size_t)arow * H_;
    const _Float16* wbase = Wm + (size_t)(h * HD_ + lanelo) * H_;

    for (int kk = 0; kk < H_; kk += 32) {
        __builtin_prefetch(xrow + kk + 128, 0, 1);      // global_prefetch_b8
        // A fragment: low lanes K 0..7 & 16..23, high lanes +8
        const int aoff = kk + hi * 8;
        const v16h a = join8(*(const v8h*)(xrow + aoff),
                             *(const v8h*)(xrow + aoff + 16));
        #pragma unroll
        for (int t = 0; t < 4; ++t) {
            // B fragment: lane = N col, K 0..7 & 16..23, high half +8
            const _Float16* wr = wbase + (size_t)(t * 16) * H_ + kk + hi * 8;
            const v16h bf = join8(*(const v8h*)wr, *(const v8h*)(wr + 16));
            acc[t] = __builtin_amdgcn_wmma_f32_16x16x32_f16(
                false, a, false, bf, (short)0, acc[t], false, false);
        }
    }

    // C/D layout: VGPR r, lanes0-15 -> M=r, lanes16-31 -> M=r+8; col = lanelo
    #pragma unroll
    for (int t = 0; t < 4; ++t) {
        const int d = t * 16 + lanelo;
        const float bias_v = bias[h * HD_ + d];
        #pragma unroll
        for (int r = 0; r < 8; ++r) {
            const int gm = mtile * 16 + r + hi * 8;
            const int b  = gm / N_;
            const int n  = gm - b * N_;
            const float val = acc[t][r] + bias_v;
            if (mat == 2) {
                vT[(((size_t)(b * NH_ + h)) * HD_ + d) * NP + n] = (_Float16)val;
            } else {
                _Float16* dst = (mat == 0) ? qws : kws;
                dst[(((size_t)(b * NH_ + h)) * NP + n) * HD_ + d] = (_Float16)val;
            }
        }
    }
}

// ---------------------------------------------------------------------------
// Fused scores + mask + softmax + entity-mask + probs@V per (b, h, qtile).
// One wave per task; 64*12*12 = 9216 waves = 1152 blocks of 8 waves.
// Probs staged through LDS (f16, stride 200) to re-layout C->A fragments.
// ---------------------------------------------------------------------------
__global__ __launch_bounds__(256) void attn_kernel(
    const _Float16* __restrict__ qws, const _Float16* __restrict__ kws,
    const _Float16* __restrict__ vT,
    const float* __restrict__ amask,        // (B,1,1,N) -> flat B*N
    const float* __restrict__ adj,          // (B,150,150,12)
    float* __restrict__ out)                // (B,N,H)
{
    __shared__ _Float16 pbuf[8][16][200];

    const int lane   = threadIdx.x & 31;
    const int wvid   = threadIdx.x >> 5;
    const int task   = blockIdx.x * 8 + wvid;       // < 9216
    const int b      = task / (NH_ * 12);
    const int rem    = task % (NH_ * 12);
    const int h      = rem / 12;
    const int qtile  = rem % 12;
    const int lanelo = lane & 15;
    const int hi     = lane >> 4;
    const size_t bh  = (size_t)(b * NH_ + h);

    // ---- scores = Q @ K^T  (12 key tiles, K-dim = HD = 2 wmma steps) ----
    v8f acc[12];
    #pragma unroll
    for (int t = 0; t < 12; ++t)
        #pragma unroll
        for (int i = 0; i < 8; ++i) acc[t][i] = 0.f;

    const _Float16* qrow = qws + (bh * NP + qtile * 16 + lanelo) * HD_;
    const _Float16* kbase = kws + (bh * NP + lanelo) * HD_;
    #pragma unroll
    for (int ks = 0; ks < 2; ++ks) {
        const int kk = ks * 32;
        const int aoff = kk + hi * 8;
        const v16h a = join8(*(const v8h*)(qrow + aoff),
                             *(const v8h*)(qrow + aoff + 16));
        #pragma unroll
        for (int t = 0; t < 12; ++t) {
            const _Float16* kr = kbase + (size_t)(t * 16) * HD_ + kk + hi * 8;
            const v16h bf = join8(*(const v8h*)kr, *(const v8h*)(kr + 16));
            acc[t] = __builtin_amdgcn_wmma_f32_16x16x32_f16(
                false, a, false, bf, (short)0, acc[t], false, false);
        }
    }

    // ---- masks + row softmax (f32), probs -> f16 LDS ----
    float am[12];
    #pragma unroll
    for (int t = 0; t < 12; ++t) {
        const int kcol = t * 16 + lanelo;
        am[t] = (kcol < N_) ? amask[b * N_ + kcol] : 0.f;
    }

    #pragma unroll
    for (int r = 0; r < 8; ++r) {
        const int q = qtile * 16 + r + hi * 8;      // query row of this element
        float x[12];
        float cmax = -3.0e38f;
        #pragma unroll
        for (int t = 0; t < 12; ++t) {
            const int kcol = t * 16 + lanelo;
            if (kcol >= N_) {
                x[t] = -1.0e30f;                    // padded key column
            } else {
                float sp;
                if (q >= MSL_ + OCR_) {
                    sp = -10000.f;                  // quadrants 7,8,9
                } else if (q < MSL_) {
                    sp = (kcol >= MSL_ + OCR_) ? 0.f : -10000.f;   // q1,q2
                } else {                            // visual query row
                    if (kcol < MSL_) sp = -10000.f; // q4
                    else if (kcol < MSL_ + OCR_) {
                        const float av = adj[(((size_t)b * OCR_ + (q - MSL_)) * OCR_
                                              + (kcol - MSL_)) * NH_ + h];
                        sp = (1.f - av) * -10000.f;
                    } else sp = 0.f;
                }
                const float comb = fminf(am[t], sp);
                cmax = fmaxf(cmax, comb);
                x[t] = acc[t][r] * 0.125f + comb;   // 1/sqrt(64)
            }
        }
        float m = x[0];
        #pragma unroll
        for (int t = 1; t < 12; ++t) m = fmaxf(m, x[t]);
        #pragma unroll
        for (int s = 8; s >= 1; s >>= 1) {
            m    = fmaxf(m,    __shfl_xor(m,    s, 32));
            cmax = fmaxf(cmax, __shfl_xor(cmax, s, 32));
        }
        float e[12], sum = 0.f;
        #pragma unroll
        for (int t = 0; t < 12; ++t) { e[t] = __expf(x[t] - m); sum += e[t]; }
        #pragma unroll
        for (int s = 8; s >= 1; s >>= 1) sum += __shfl_xor(sum, s, 32);
        const float ent   = (cmax + 10000.f) * 1.0e-4f;   // {0,1}
        const float scale = ent / sum;
        #pragma unroll
        for (int t = 0; t < 12; ++t)
            pbuf[wvid][r + hi * 8][t * 16 + lanelo] = (_Float16)(e[t] * scale);
    }

    __syncthreads();   // uniform across the block; orders LDS write->read

    // ---- ctx = probs @ V   (K = 192 -> 6 wmma steps, 4 HD tiles) ----
    v8f cacc[4];
    #pragma unroll
    for (int t = 0; t < 4; ++t)
        #pragma unroll
        for (int i = 0; i < 8; ++i) cacc[t][i] = 0.f;

    const _Float16* vbase = vT + (bh * HD_ + lanelo) * NP;
    for (int kk = 0; kk < NP; kk += 32) {
        const int aoff = kk + hi * 8;
        const v16h a = join8(*(const v8h*)&pbuf[wvid][lanelo][aoff],
                             *(const v8h*)&pbuf[wvid][lanelo][aoff + 16]);
        #pragma unroll
        for (int t = 0; t < 4; ++t) {
            const _Float16* vr = vbase + (size_t)(t * 16) * NP + kk + hi * 8;
            const v16h bf = join8(*(const v8h*)vr, *(const v8h*)(vr + 16));
            cacc[t] = __builtin_amdgcn_wmma_f32_16x16x32_f16(
                false, a, false, bf, (short)0, cacc[t], false, false);
        }
    }

    #pragma unroll
    for (int t = 0; t < 4; ++t) {
        const int d = t * 16 + lanelo;
        #pragma unroll
        for (int r = 0; r < 8; ++r) {
            const int n = qtile * 16 + r + hi * 8;
            if (n < N_)
                out[((size_t)b * N_ + n) * H_ + h * HD_ + d] = cacc[t][r];
        }
    }
}

// ---------------------------------------------------------------------------
extern "C" void kernel_launch(void* const* d_in, const int* in_sizes, int n_in,
                              void* d_out, int out_size, void* d_ws, size_t ws_size,
                              hipStream_t stream) {
    const float* X     = (const float*)d_in[0];   // hidden_states (B,N,H)
    const float* amask = (const float*)d_in[1];   // attention_mask (B,1,1,N)
    const float* adj   = (const float*)d_in[2];   // spatial_adj (B,150,150,12)
    const float* Wq    = (const float*)d_in[3];
    const float* bq    = (const float*)d_in[4];
    const float* Wk    = (const float*)d_in[5];
    const float* bk    = (const float*)d_in[6];
    const float* Wv    = (const float*)d_in[7];
    const float* bv    = (const float*)d_in[8];
    float* out = (float*)d_out;

    const size_t QK_ELEMS = (size_t)B_ * NH_ * NP * HD_;   // 9,437,184 halves
    const size_t XH_ELEMS = (size_t)B_ * N_ * H_;          // 8,945,664
    const size_t W_ELEMS  = (size_t)H_ * H_;               // 589,824

    _Float16* qws = (_Float16*)d_ws;
    _Float16* kws = qws + QK_ELEMS;
    _Float16* vT  = kws + QK_ELEMS;
    _Float16* Xh  = vT  + QK_ELEMS;
    _Float16* Wh  = Xh  + XH_ELEMS;                        // (3,768,768)

    zero_pad_kernel<<<dim3(1920), dim3(256), 0, stream>>>(qws, kws, vT);

    // one-shot f32 -> f16 conversions (4 elems/thread)
    cvt_f32_to_f16_kernel<<<dim3((int)(XH_ELEMS / 1024)), dim3(256), 0, stream>>>(
        X, Xh, (int)(XH_ELEMS / 4));
    cvt_f32_to_f16_kernel<<<dim3((int)(W_ELEMS / 1024)), dim3(256), 0, stream>>>(
        Wq, Wh, (int)(W_ELEMS / 4));
    cvt_f32_to_f16_kernel<<<dim3((int)(W_ELEMS / 1024)), dim3(256), 0, stream>>>(
        Wk, Wh + W_ELEMS, (int)(W_ELEMS / 4));
    cvt_f32_to_f16_kernel<<<dim3((int)(W_ELEMS / 1024)), dim3(256), 0, stream>>>(
        Wv, Wh + 2 * W_ELEMS, (int)(W_ELEMS / 4));

    qkv_proj_kernel<<<dim3(3276), dim3(256), 0, stream>>>(
        Xh, Wh, bq, bk, bv, qws, kws, vT);
    attn_kernel<<<dim3(1152), dim3(256), 0, stream>>>(
        qws, kws, vT, amask, adj, out);
}
